// PPGN_eff_23888608100575
// MI455X (gfx1250) — matrix operand order repaired and checked by reading, not compiled, not run on hardware
//
#include <hip/hip_runtime.h>
#include <hip/hip_bf16.h>

// ---------------------------------------------------------------------------
// PPGN on MI455X (gfx1250, wave32).  Heavy math via v_wmma_f32_16x16x32_bf16
// (bf16 operands packed in-register with v_perm_b32, fp32 accumulation).
// ---------------------------------------------------------------------------

typedef __attribute__((ext_vector_type(16))) __bf16 v16bf;
typedef __attribute__((ext_vector_type(8)))  float  v8f;

union V16U { v16bf v; unsigned short u[16]; unsigned d[8]; };

// pack two fp32 -> one dword of two bf16 (truncation) with a single v_perm_b32
__device__ __forceinline__ unsigned pk2(float lo, float hi) {
  return __builtin_amdgcn_perm(__float_as_uint(hi), __float_as_uint(lo), 0x07060302u);
}
__device__ __forceinline__ unsigned short f2bf_rne(float f) {
  unsigned u = __float_as_uint(f);
  unsigned r = u + 0x7FFFu + ((u >> 16) & 1u);
  return (unsigned short)(r >> 16);
}
__device__ __forceinline__ unsigned short f2bf_tr(float f) {
  return (unsigned short)(__float_as_uint(f) >> 16);
}

#define FLAG_RELU 1
#define FLAG_BN   2

// ---------------------------------------------------------------------------
// Weight packer: fp32 W -> bf16 (RNE) in WMMA B-operand layout.
// B (32x16 bf16): lane l holds column n=(l&15); halves e=0..7 hold
// k = (l>>4)*8 + e, e=8..15 hold k = (l>>4)*8 + e + 8.  Tiles stored [ktile][ntile][lane][e], 512 halves/tile.
// k maps through two zero-padded segments (for the concat skip-conv).
// ---------------------------------------------------------------------------
__global__ void pack_w_kernel(const float* __restrict__ W, int ldw, int trans,
                              int K1act, int K1pad, int K2act,
                              __bf16* __restrict__ out)
{
  const int ntiles = 4;                       // all GEMMs here have Nout = 64
  int ktile = blockIdx.x / ntiles;
  int ntile = blockIdx.x % ntiles;
  int lane  = threadIdx.x;                    // 32 threads = 1 wave
  unsigned short* dst = (unsigned short*)out + (size_t)blockIdx.x * 512 + lane * 16;
#pragma unroll
  for (int e = 0; e < 16; ++e) {
    int k = ktile * 32 + (lane >> 4) * 8 + (e < 8 ? e : 16 + (e - 8));
    int n = ntile * 16 + (lane & 15);
    int r;
    if (k < K1pad) r = (k < K1act) ? k : -1;
    else { int k2 = k - K1pad; r = (k2 < K2act) ? (K1act + k2) : -1; }
    float f = 0.f;
    if (r >= 0) f = trans ? W[(size_t)n * ldw + r] : W[(size_t)r * ldw + n];
    dst[e] = f2bf_rne(f);
  }
}

// ---------------------------------------------------------------------------
// Generic GEMM  Y[M,64] = relu/bn( [X1 | X2][M, K1+K2] * Wpacked + bias )
// 256 threads = 8 waves; wave w -> (mtile=w>>2, ntile=w&3); block = 32 rows.
// A (16x32 bf16): lane l -> row (l&15); k groups {rowhalf*8, rowhalf*8+16}
// are 8 contiguous fp32 each -> aligned float4 pairs -> v_perm packing.
// Constraints: K1 multiple of 8 (tail tile guarded); K2 multiple of 32 or 0;
// X1/X2 zero-padded up to their ld so full-tile loads are branch-free.
// ---------------------------------------------------------------------------
__global__ __launch_bounds__(256)
void gemm_bf16_kernel(const float* __restrict__ X1, int ldX1, int K1,
                      const float* __restrict__ X2, int ldX2, int K2,
                      const __bf16* __restrict__ Wp,
                      const float* __restrict__ bias,
                      const float* __restrict__ bng, const float* __restrict__ bnb,
                      const float* __restrict__ bnm, const float* __restrict__ bnv,
                      float* __restrict__ Y, int ldY, int flags)
{
  const int lane    = threadIdx.x & 31;
  const int wave    = threadIdx.x >> 5;
  const int ntile   = wave & 3;
  const int mtile   = wave >> 2;            // 0..1
  const int rowhalf = lane >> 4;
  const int lidx    = lane & 15;
  const int row     = blockIdx.x * 32 + mtile * 16 + lidx;

  const unsigned short* pW =
      (const unsigned short*)Wp + (size_t)ntile * 512 + lane * 16;
  v8f acc = {};

  // ---- phase 1: full 32-wide K tiles from X1 (branch-free) ----
  const float* pA = X1 + (size_t)row * ldX1 + rowhalf * 8;
  const int nt1 = K1 >> 5;
  for (int kt = 0; kt < nt1; ++kt) {
    V16U a;
    float4 f0 = *(const float4*)(pA);
    float4 f1 = *(const float4*)(pA + 4);
    float4 g0 = *(const float4*)(pA + 16);
    float4 g1 = *(const float4*)(pA + 20);
    a.d[0] = pk2(f0.x, f0.y); a.d[1] = pk2(f0.z, f0.w);
    a.d[2] = pk2(f1.x, f1.y); a.d[3] = pk2(f1.z, f1.w);
    a.d[4] = pk2(g0.x, g0.y); a.d[5] = pk2(g0.z, g0.w);
    a.d[6] = pk2(g1.x, g1.y); a.d[7] = pk2(g1.z, g1.w);
    v16bf b = *(const v16bf*)pW;
    acc = __builtin_amdgcn_wmma_f32_16x16x32_bf16(
        false, a.v, false, b, (short)0, acc, false, false);
    pA += 32; pW += 2048;
  }

  // ---- phase 2: partial X1 tail tile (only the K=1800 GEMM takes this) ----
  if (K1 & 31) {
    V16U a;
    auto guard = [&](unsigned* d, int kb) {
      if (kb + 8 <= K1) {
        const float* p = X1 + (size_t)row * ldX1 + kb;
        float4 f0 = *(const float4*)(p);
        float4 f1 = *(const float4*)(p + 4);
        d[0] = pk2(f0.x, f0.y); d[1] = pk2(f0.z, f0.w);
        d[2] = pk2(f1.x, f1.y); d[3] = pk2(f1.z, f1.w);
      } else { d[0] = d[1] = d[2] = d[3] = 0u; }
    };
    int kb = nt1 * 32 + rowhalf * 8;
    guard(a.d + 0, kb);
    guard(a.d + 4, kb + 16);
    v16bf b = *(const v16bf*)pW;
    acc = __builtin_amdgcn_wmma_f32_16x16x32_bf16(
        false, a.v, false, b, (short)0, acc, false, false);
    pW += 2048;
  }

  // ---- phase 3: full 32-wide K tiles from X2 (concat segment) ----
  if (K2) {
    const float* pA2 = X2 + (size_t)row * ldX2 + rowhalf * 8;
    const int nt2 = K2 >> 5;
    for (int kt = 0; kt < nt2; ++kt) {
      V16U a;
      float4 f0 = *(const float4*)(pA2);
      float4 f1 = *(const float4*)(pA2 + 4);
      float4 g0 = *(const float4*)(pA2 + 16);
      float4 g1 = *(const float4*)(pA2 + 20);
      a.d[0] = pk2(f0.x, f0.y); a.d[1] = pk2(f0.z, f0.w);
      a.d[2] = pk2(f1.x, f1.y); a.d[3] = pk2(f1.z, f1.w);
      a.d[4] = pk2(g0.x, g0.y); a.d[5] = pk2(g0.z, g0.w);
      a.d[6] = pk2(g1.x, g1.y); a.d[7] = pk2(g1.z, g1.w);
      v16bf b = *(const v16bf*)pW;
      acc = __builtin_amdgcn_wmma_f32_16x16x32_bf16(
          false, a.v, false, b, (short)0, acc, false, false);
      pA2 += 32; pW += 2048;
    }
  }

  // ---- epilogue: bias (+BN) (+ReLU) ----
  const int col = ntile * 16 + lidx;
  float bs = bias[col], sc = 1.f, sh = 0.f;
  if (flags & FLAG_BN) {
    float rs = rsqrtf(bnv[col] + 1e-5f);
    sc = bng[col] * rs;
    sh = bnb[col] - bnm[col] * sc;
  }
#pragma unroll
  for (int v = 0; v < 8; ++v) {
    int orow = blockIdx.x * 32 + mtile * 16 + rowhalf * 8 + v;
    float val = (acc[v] + bs) * sc + sh;
    if (flags & FLAG_RELU) val = fmaxf(val, 0.f);
    Y[(size_t)orow * ldY + col] = val;
  }
}

// ---------------------------------------------------------------------------
// to_dense_adj scatter: z0[b, r, c, :] += [1, edge_emb];  layout [B,N,N,96].
// ---------------------------------------------------------------------------
__global__ void scatter_edges_kernel(const int* __restrict__ ei,
                                     const float* __restrict__ emb,
                                     float* __restrict__ z0, int E)
{
  int t = blockIdx.x * 256 + threadIdx.x;
  int e = t >> 6, c = t & 63;
  if (e >= E) return;
  int s = ei[e], d = ei[E + e];
  int gid = s >> 6, r = s & 63, cc = d & 63;
  size_t base = (((size_t)gid * 64 + r) * 64 + cc) * 96;
  atomicAdd(&z0[base + 1 + c], emb[(size_t)e * 64 + c]);
  if (c == 0) atomicAdd(&z0[base], 1.0f);
}

// ---------------------------------------------------------------------------
// Batched einsum bcij,bcjk->bcik : one workgroup per (b,c); m1 staged to LDS
// row-major, m2 staged transposed, both bf16 -> WMMA operands become two
// contiguous ds_load_b128 each.  16 output tiles, 2 per wave.
// ---------------------------------------------------------------------------
__global__ __launch_bounds__(256)
void bmm_kernel(const float* __restrict__ M1, const float* __restrict__ M2,
                float* __restrict__ OUT)
{
  __shared__ alignas(16) unsigned short s1[64 * 64];   // s1[i*64+j]  = m1[i][j]
  __shared__ alignas(16) unsigned short s2[64 * 64];   // s2[n*64+k]  = m2[k][n]
  int b = blockIdx.x >> 6;
  int c = blockIdx.x & 63;
  size_t base = (size_t)b * 64 * 64 * 64 + c;
  for (int idx = threadIdx.x; idx < 4096; idx += 256) {
    int i = idx >> 6, j = idx & 63;
    s1[idx]          = f2bf_tr(M1[base + (size_t)idx * 64]);
    s2[j * 64 + i]   = f2bf_tr(M2[base + (size_t)idx * 64]);
  }
  __syncthreads();

  int lane = threadIdx.x & 31, wave = threadIdx.x >> 5;
  int rowhalf = lane >> 4, lidx = lane & 15;
  for (int t = wave * 2; t < wave * 2 + 2; ++t) {
    int ti = t >> 2, tj = t & 3;
    v8f acc = {};
#pragma unroll
    for (int kt = 0; kt < 2; ++kt) {
      V16U a, bb;
      const unsigned short* pa = s1 + (ti * 16 + lidx) * 64 + kt * 32 + rowhalf * 8;
      uint4 a0 = *(const uint4*)(pa);
      uint4 a1 = *(const uint4*)(pa + 16);
      a.d[0] = a0.x; a.d[1] = a0.y; a.d[2] = a0.z; a.d[3] = a0.w;
      a.d[4] = a1.x; a.d[5] = a1.y; a.d[6] = a1.z; a.d[7] = a1.w;
      const unsigned short* pb = s2 + (tj * 16 + lidx) * 64 + kt * 32 + rowhalf * 8;
      uint4 b0 = *(const uint4*)(pb);
      uint4 b1 = *(const uint4*)(pb + 16);
      bb.d[0] = b0.x; bb.d[1] = b0.y; bb.d[2] = b0.z; bb.d[3] = b0.w;
      bb.d[4] = b1.x; bb.d[5] = b1.y; bb.d[6] = b1.z; bb.d[7] = b1.w;
      acc = __builtin_amdgcn_wmma_f32_16x16x32_bf16(
          false, a.v, false, bb.v, (short)0, acc, false, false);
    }
#pragma unroll
    for (int v = 0; v < 8; ++v) {
      int i  = ti * 16 + rowhalf * 8 + v;
      int k2 = tj * 16 + lidx;
      OUT[base + ((size_t)i * 64 + k2) * 64] = acc[v];
    }
  }
}

// ---------------------------------------------------------------------------
// diag_offdiag pooling: P[b,n,:] = [diag(64) | rowsum+colsum-2*diag (64)].
// ---------------------------------------------------------------------------
__global__ void pool_kernel(const float* __restrict__ Z, float* __restrict__ P)
{
  int bn = blockIdx.x, c = threadIdx.x;       // 64 threads
  int b = bn >> 6, n = bn & 63;
  size_t zb = (size_t)b * 64 * 64 * 64;
  float diag = Z[zb + ((size_t)n * 64 + n) * 64 + c];
  float rs = 0.f, cs = 0.f;
  for (int j = 0; j < 64; ++j) {
    rs += Z[zb + ((size_t)n * 64 + j) * 64 + c];
    cs += Z[zb + ((size_t)j * 64 + n) * 64 + c];
  }
  P[(size_t)bn * 128 + c]      = diag;
  P[(size_t)bn * 128 + 64 + c] = rs + cs - 2.f * diag;
}

__global__ void fc2_kernel(const float* __restrict__ X, const float* __restrict__ W,
                           const float* __restrict__ Bv, float* __restrict__ out, int M)
{
  int t = blockIdx.x * 256 + threadIdx.x;
  if (t >= M) return;
  float s = Bv[0];
  for (int c = 0; c < 64; ++c) s += X[(size_t)t * 64 + c] * W[c];
  out[t] = s;
}

// ---------------------------------------------------------------------------
// Host side
// ---------------------------------------------------------------------------
extern "C" void kernel_launch(void* const* d_in, const int* in_sizes, int n_in,
                              void* d_out, int out_size, void* d_ws, size_t ws_size,
                              hipStream_t stream)
{
  (void)in_sizes; (void)n_in; (void)out_size; (void)ws_size;
  const int Bg = 128, Nn = 64, E = 65536, EPOS = 1800;
  const size_t M = (size_t)Bg * Nn * Nn;            // 524288

  // ---- input leaves (setup_inputs dict insertion order) ----
  const float* edge_pos = (const float*)d_in[0];
  const int*   edge_idx = (const int*)d_in[1];
  // d_in[2] = batch (unused: all graphs are full)
  const float* ze_w1 = (const float*)d_in[3];
  const float* ze_b1 = (const float*)d_in[4];
  const float* ze_g1 = (const float*)d_in[5];
  const float* ze_be1 = (const float*)d_in[6];
  const float* ze_m1 = (const float*)d_in[7];
  const float* ze_v1 = (const float*)d_in[8];
  const float* ze_w2 = (const float*)d_in[9];
  const float* ze_b2 = (const float*)d_in[10];
  const float* ze_g2 = (const float*)d_in[11];
  const float* ze_be2 = (const float*)d_in[12];
  const float* ze_m2 = (const float*)d_in[13];
  const float* ze_v2 = (const float*)d_in[14];
  const float* rb[4][10];
  for (int b = 0; b < 4; ++b)
    for (int j = 0; j < 10; ++j)
      rb[b][j] = (const float*)d_in[15 + b * 10 + j];
  const float* fc1_w = (const float*)d_in[55];
  const float* fc1_b = (const float*)d_in[56];
  const float* fc2_w = (const float*)d_in[57];
  const float* fc2_b = (const float*)d_in[58];

  // ---- workspace layout ----
  char* ws = (char*)d_ws;
  const size_t SZ_Z0  = M * 96 * 4;        // channels-last z0, C padded 66->96
  const size_t SZ_BUF = M * 64 * 4;
  size_t off = 0;
  float* Z0   = (float*)(ws + off); off += SZ_Z0;
  float* BA   = (float*)(ws + off); off += SZ_BUF;
  float* BB   = (float*)(ws + off); off += SZ_BUF;
  float* BC   = (float*)(ws + off); off += SZ_BUF;
  float* BD   = (float*)(ws + off); off += SZ_BUF;
  float* H    = (float*)(ws + off); off += (size_t)E * 64 * 4;
  float* EMB  = (float*)(ws + off); off += (size_t)E * 64 * 4;
  float* POOL = (float*)(ws + off); off += (size_t)8192 * 128 * 4;
  float* FC1O = (float*)(ws + off); off += (size_t)8192 * 64 * 4;
  char* pk = ws + off;
  __bf16* PK_W1 = (__bf16*)(pk);               // 57 ktiles
  __bf16* PK_W2 = (__bf16*)(pk + 233472);      // 2 ktiles
  __bf16* PK_RB[4][5];
  {
    size_t p = 241664;
    for (int b = 0; b < 4; ++b)
      for (int j = 0; j < 5; ++j) { PK_RB[b][j] = (__bf16*)(pk + p); p += 20480; }
  }
  __bf16* PK_FC1 = (__bf16*)(pk + 241664 + 20 * 20480);

  auto pack = [&](const float* W, int ldw, int trans, int K1act, int K1pad,
                  int K2act, __bf16* out) {
    int ktiles = (K1pad + K2act + 31) / 32;
    pack_w_kernel<<<ktiles * 4, 32, 0, stream>>>(W, ldw, trans, K1act, K1pad, K2act, out);
  };
  auto gemm = [&](const float* X1, int ldX1, int K1, const float* X2, int ldX2,
                  int K2, const __bf16* Wp, const float* bias, const float* g,
                  const float* be, const float* mm, const float* vv,
                  float* Y, int ldY, size_t Mrows, int flags) {
    gemm_bf16_kernel<<<(int)(Mrows / 32), 256, 0, stream>>>(
        X1, ldX1, K1, X2, ldX2, K2, Wp, bias, g, be, mm, vv, Y, ldY, flags);
  };

  // ---- pack all weights (bf16, WMMA-B layout) ----
  pack(ze_w1, 64, 0, EPOS, EPOS, 0, PK_W1);         // [1800,64]  K x N
  pack(ze_w2, 64, 0, 64, 64, 0, PK_W2);             // [64,64]
  for (int b = 0; b < 4; ++b) {
    int Cin  = (b == 0) ? 66 : 64;
    int Cpad = (b == 0) ? 96 : 64;
    pack(rb[b][0], Cin, 1, Cin, Cpad, 0, PK_RB[b][0]);       // mlp1 w0 [64,Cin]
    pack(rb[b][2], 64, 1, 64, 64, 0, PK_RB[b][1]);           // mlp1 w1
    pack(rb[b][4], Cin, 1, Cin, Cpad, 0, PK_RB[b][2]);       // mlp2 w0
    pack(rb[b][6], 64, 1, 64, 64, 0, PK_RB[b][3]);           // mlp2 w1
    pack(rb[b][8], Cin + 64, 1, Cin, Cpad, 64, PK_RB[b][4]); // skip [64, Cin+64]
  }
  pack(fc1_w, 64, 0, 128, 128, 0, PK_FC1);          // [128,64]

  // ---- z_embedding: two fused GEMM+BN+ReLU ----
  gemm(edge_pos, EPOS, EPOS, nullptr, 0, 0, PK_W1, ze_b1, ze_g1, ze_be1, ze_m1,
       ze_v1, H, 64, E, FLAG_BN | FLAG_RELU);
  gemm(H, 64, 64, nullptr, 0, 0, PK_W2, ze_b2, ze_g2, ze_be2, ze_m2, ze_v2,
       EMB, 64, E, FLAG_BN | FLAG_RELU);

  // ---- dense adjacency (zero then scatter-add) ----
  hipMemsetAsync(Z0, 0, SZ_Z0, stream);
  scatter_edges_kernel<<<(E * 64) / 256, 256, 0, stream>>>(edge_idx, EMB, Z0, E);

  // ---- 4 regular blocks ----
  const float* x = Z0;  int ldx = 96, Kx = 96;
  for (int b = 0; b < 4; ++b) {
    gemm(x, ldx, Kx, nullptr, 0, 0, PK_RB[b][0], rb[b][1], 0, 0, 0, 0, BA, 64, M, FLAG_RELU);
    gemm(BA, 64, 64, nullptr, 0, 0, PK_RB[b][1], rb[b][3], 0, 0, 0, 0, BB, 64, M, FLAG_RELU);
    gemm(x, ldx, Kx, nullptr, 0, 0, PK_RB[b][2], rb[b][5], 0, 0, 0, 0, BA, 64, M, FLAG_RELU);
    gemm(BA, 64, 64, nullptr, 0, 0, PK_RB[b][3], rb[b][7], 0, 0, 0, 0, BC, 64, M, FLAG_RELU);
    bmm_kernel<<<Bg * 64, 256, 0, stream>>>(BB, BC, BA);       // mult -> BA
    float* Y = (b & 1) ? Z0 : BD;                              // ping-pong target
    gemm(x, ldx, Kx, BA, 64, 64, PK_RB[b][4], rb[b][9], 0, 0, 0, 0, Y, 64, M, 0);
    x = Y; ldx = 64; Kx = 64;
  }
  // after 4 blocks the result lives in Z0 (block 3 wrote Z0), ld 64

  // ---- pooling + MLP head ----
  pool_kernel<<<Bg * Nn, 64, 0, stream>>>(x, POOL);
  gemm(POOL, 128, 128, nullptr, 0, 0, PK_FC1, fc1_b, 0, 0, 0, 0, FC1O, 64, 8192, FLAG_RELU);
  fc2_kernel<<<8192 / 256, 256, 0, stream>>>(FC1O, fc2_w, fc2_b, (float*)d_out, 8192);
}
